// StaticTernaryLinear_73280732005201
// MI455X (gfx1250) — 8-bit matrix operand order repaired and checked by reading, compiled, not run on hardware
//
#include <hip/hip_runtime.h>

typedef __attribute__((ext_vector_type(8))) int v8i;
typedef __attribute__((ext_vector_type(4))) int v4i;

#define M_DIM 8192
#define N_DIM 16384
#define K_DIM 4096
#define PITCH 80

#if __has_builtin(__builtin_amdgcn_global_load_async_to_lds_b128) && \
    __has_builtin(__builtin_amdgcn_s_wait_asynccnt)
#define USE_ASYNC 1
#else
#define USE_ASYNC 0
#endif

typedef __attribute__((address_space(1))) v4i gv4i;
typedef __attribute__((address_space(3))) v4i lv4i;

__global__ void quant_kernel(const float* __restrict__ x,
                             signed char* __restrict__ qx,
                             float* __restrict__ act_scale) {
    __shared__ float red[256];
    const int token = blockIdx.x;
    const int t = threadIdx.x;
    const float4* row4 = (const float4*)(x + (size_t)token * K_DIM);

    float4 v[4];
    float amax = 0.0f;
#pragma unroll
    for (int i = 0; i < 4; ++i) {
        v[i] = row4[t + i * 256];
        amax = fmaxf(amax, fmaxf(fmaxf(fabsf(v[i].x), fabsf(v[i].y)),
                                 fmaxf(fabsf(v[i].z), fabsf(v[i].w))));
    }
    red[t] = amax;
    __syncthreads();
    for (int s = 128; s > 0; s >>= 1) {
        if (t < s) red[t] = fmaxf(red[t], red[t + s]);
        __syncthreads();
    }
    const float scale = fmaxf(red[0], 1e-5f) * (1.0f / 127.0f);
    const float inv = 1.0f / scale;
    if (t == 0) act_scale[token] = scale;

    int* qrow = (int*)(qx + (size_t)token * K_DIM);
#pragma unroll
    for (int i = 0; i < 4; ++i) {
        float4 f = v[i];
        int b0 = (int)rintf(f.x * inv);
        int b1 = (int)rintf(f.y * inv);
        int b2 = (int)rintf(f.z * inv);
        int b3 = (int)rintf(f.w * inv);
        b0 = min(127, max(-128, b0));
        b1 = min(127, max(-128, b1));
        b2 = min(127, max(-128, b2));
        b3 = min(127, max(-128, b3));
        qrow[t + i * 256] = (b0 & 0xff) | ((b1 & 0xff) << 8) |
                            ((b2 & 0xff) << 16) | (b3 << 24);
    }
}

__global__ void pack_w(const int* __restrict__ w, signed char* __restrict__ wq) {
    size_t i = ((size_t)blockIdx.x * 256 + threadIdx.x) * 4;
    int4 v = *(const int4*)(w + i);
    *(int*)(wq + i) = (v.x & 0xff) | ((v.y & 0xff) << 8) |
                      ((v.z & 0xff) << 16) | (v.w << 24);
}

__device__ __forceinline__ void cp16(const signed char* g, signed char* l) {
#if USE_ASYNC
    __builtin_amdgcn_global_load_async_to_lds_b128((gv4i*)g, (lv4i*)l, 0, 0);
#else
    *(int4*)l = *(const int4*)g;
#endif
}

__device__ __forceinline__ void stage_tiles(const signed char* ga,
                                            const signed char* gb, int kc,
                                            signed char* bufA, signed char* bufB,
                                            int crow, int ccol) {
#pragma unroll
    for (int r = 0; r < 2; ++r) {
        int row = crow + r * 64;
        cp16(ga + (size_t)row * K_DIM + kc + ccol, bufA + row * PITCH + ccol);
    }
#pragma unroll
    for (int r = 0; r < 4; ++r) {
        int row = crow + r * 64;
        cp16(gb + (size_t)row * K_DIM + kc + ccol, bufB + row * PITCH + ccol);
    }
}

__device__ __forceinline__ v8i load_a_frag(const signed char* p) {
    int2 a = *(const int2*)(p);
    int2 b = *(const int2*)(p + 16);
    int2 c = *(const int2*)(p + 32);
    int2 d = *(const int2*)(p + 48);
    v8i r;
    r[0] = a.x; r[1] = a.y; r[2] = b.x; r[3] = b.y;
    r[4] = c.x; r[5] = c.y; r[6] = d.x; r[7] = d.y;
    return r;
}
__device__ __forceinline__ v8i load_b_frag(const signed char* p) {
    int2 a = *(const int2*)(p);
    int2 b = *(const int2*)(p + 16);
    int2 c = *(const int2*)(p + 32);
    int2 d = *(const int2*)(p + 48);
    v8i r;
    r[0] = a.x; r[1] = a.y; r[2] = b.x; r[3] = b.y;
    r[4] = c.x; r[5] = c.y; r[6] = d.x; r[7] = d.y;
    return r;
}

__global__ __launch_bounds__(256, 1) void ternary_gemm(
    const signed char* __restrict__ qx,
    const signed char* __restrict__ wq,
    const float* __restrict__ act_scale,
    const float* __restrict__ wscale,
    const float* __restrict__ bias,
    float* __restrict__ out) {
    __shared__ signed char lsA[2][128 * PITCH];
    __shared__ signed char lsB[2][256 * PITCH];

    const int t    = threadIdx.x;
    const int wave = t >> 5;
    const int lane = t & 31;
    const int m0 = blockIdx.y * 128;
    const int n0 = blockIdx.x * 256;
    const int wm = (wave & 1) * 64;
    const int wn = (wave >> 1) * 64;

    const int lrow = lane & 15;
    const int lkg  = lane >> 4;

    const int crow = t >> 2;
    const int ccol = (t & 3) * 16;

    v8i acc[4][4];
    const v8i vzero = {0, 0, 0, 0, 0, 0, 0, 0};
#pragma unroll
    for (int i = 0; i < 4; ++i)
#pragma unroll
        for (int j = 0; j < 4; ++j) acc[i][j] = vzero;

    const signed char* ga = qx + (size_t)m0 * K_DIM;
    const signed char* gb = wq + (size_t)n0 * K_DIM;

    stage_tiles(ga, gb, 0, lsA[0], lsB[0], crow, ccol);

    int buf = 0;
    for (int kc = 0; kc < K_DIM; kc += 64, buf ^= 1) {
        if (kc + 64 < K_DIM) {
            stage_tiles(ga, gb, kc + 64, lsA[buf ^ 1], lsB[buf ^ 1], crow, ccol);
#if USE_ASYNC
            __builtin_amdgcn_s_wait_asynccnt(6);
#endif
        } else {
#if USE_ASYNC
            __builtin_amdgcn_s_wait_asynccnt(0);
#endif
        }
        __syncthreads();

        v8i afrag[4], bfrag[4];
#pragma unroll
        for (int i = 0; i < 4; ++i)
            afrag[i] =
                load_a_frag(lsA[buf] + (wm + i * 16 + lrow) * PITCH + lkg * 8);
#pragma unroll
        for (int j = 0; j < 4; ++j)
            bfrag[j] =
                load_b_frag(lsB[buf] + (wn + j * 16 + lrow) * PITCH + lkg * 8);

#pragma unroll
        for (int i = 0; i < 4; ++i)
#pragma unroll
            for (int j = 0; j < 4; ++j)
                acc[i][j] = __builtin_amdgcn_wmma_i32_16x16x64_iu8(
                     true, afrag[i],
                     true, bfrag[j],
                    acc[i][j],  false,  false);

        __syncthreads();
    }

    const float ws = wscale[0];
#pragma unroll
    for (int i = 0; i < 4; ++i) {
        const int mb = m0 + wm + i * 16 + (lane >> 4) * 8;
#pragma unroll
        for (int j = 0; j < 4; ++j) {
            const int n = n0 + wn + j * 16 + (lane & 15);
            const float b = bias[n];
#pragma unroll
            for (int r = 0; r < 8; ++r) {
                const int m = mb + r;
                out[(size_t)m * N_DIM + n] =
                    (float)acc[i][j][r] * act_scale[m] * ws + b;
            }
        }
    }
}

extern "C" void kernel_launch(void* const* d_in, const int* in_sizes, int n_in,
                              void* d_out, int out_size, void* d_ws, size_t ws_size,
                              hipStream_t stream) {
    const float* x      = (const float*)d_in[0];
    const int*   w      = (const int*)d_in[1];
    const float* wscale = (const float*)d_in[2];
    const float* bias   = (const float*)d_in[3];
    float* out = (float*)d_out;

    signed char* qx = (signed char*)d_ws;
    signed char* wq = qx + (size_t)M_DIM * K_DIM;
    float* ascale   = (float*)(wq + (size_t)N_DIM * K_DIM);

    quant_kernel<<<M_DIM, 256, 0, stream>>>(x, qx, ascale);
    pack_w<<<(N_DIM * (size_t)K_DIM) / 1024, 256, 0, stream>>>(w, wq);

    dim3 grid(N_DIM / 256, M_DIM / 128);
    ternary_gemm<<<grid, 256, 0, stream>>>(qx, wq, ascale, wscale, bias, out);
}
